// Cross_Attention_87986700026081
// MI455X (gfx1250) — matrix operand order repaired and checked by reading, not compiled, not run on hardware
//
#include <hip/hip_runtime.h>
#include <hip/hip_bf16.h>

typedef __attribute__((ext_vector_type(16))) _Float16 v16h;
typedef __attribute__((ext_vector_type(8)))  _Float16 v8h;
typedef __attribute__((ext_vector_type(4)))  _Float16 v4h;
typedef __attribute__((ext_vector_type(8)))  float    v8f;
typedef __attribute__((ext_vector_type(4)))  float    v4f;
typedef __attribute__((ext_vector_type(4)))  int      v4i;

#define BB 16
#define CC 256
#define NN 9216          // 96*96
#define KSPLIT 4
#define KSEG (NN / KSPLIT)   // 2304 = 36 * 64

// ---- gfx1250 async global->LDS support (guarded; falls back cleanly) ------
#if defined(__has_builtin)
#  if __has_builtin(__builtin_amdgcn_global_load_async_to_lds_b128)
#    define HAVE_ASYNC_LDS 1
#  endif
#  if __has_builtin(__builtin_amdgcn_s_wait_asynccnt)
#    define HAVE_WAIT_ASYNC 1
#  endif
#endif
#ifndef HAVE_ASYNC_LDS
#  define HAVE_ASYNC_LDS 0
#endif
#ifndef HAVE_WAIT_ASYNC
#  define HAVE_WAIT_ASYNC 0
#endif

#if HAVE_ASYNC_LDS
typedef __attribute__((address_space(1))) v4i gv4i;   // global (AS1)
typedef __attribute__((address_space(3))) v4i lv4i;   // LDS (AS3)
#endif

__device__ __forceinline__ void wait_async_zero() {
#if HAVE_ASYNC_LDS
#  if HAVE_WAIT_ASYNC
  __builtin_amdgcn_s_wait_asynccnt(0);
#  else
  asm volatile("s_wait_asynccnt 0" ::: "memory");
#  endif
#endif
}

// ---------------------------------------------------------------------------
// WMMA fragment loaders from LDS (wave32, 16-bit data, 16x16x32 shape).
// A-matrix 16x32 (ISA 7.12.2): lanes 0-15 row=M=lane, elems 0..7 = K0..7,
// elems 8..15 = K16..23; lanes 16-31 same rows, K shifted by +8.
// B-matrix 32x16: lane = column N, same K order as A (elems 0..7 = K0..7, 8..15 = K16..23; lanes 16-31 K +8).
// Rows in LDS are stored major-along-K with 16B-aligned padded strides.
// ---------------------------------------------------------------------------
__device__ __forceinline__ v16h load_a_frag(const _Float16* __restrict__ row_base,
                                            int lane) {
  const _Float16* p = row_base + (((lane >> 4) & 1) << 3);   // +0 or +8 halves
  v8h lo = *(const v8h*)(p);
  v8h hi = *(const v8h*)(p + 16);
  return __builtin_shufflevector(lo, hi, 0,1,2,3,4,5,6,7,8,9,10,11,12,13,14,15);
}

__device__ __forceinline__ v16h load_b_frag(const _Float16* __restrict__ row_base,
                                            int lane) {
  const _Float16* p = row_base + (((lane >> 4) & 1) << 3);   // +0 or +8 halves
  v8h lo = *(const v8h*)(p);
  v8h hi = *(const v8h*)(p + 16);
  return __builtin_shufflevector(lo, hi, 0,1,2,3,4,5,6,7,8,9,10,11,12,13,14,15);
}

// ---------------------------------------------------------------------------
// Kernel 1: energy partials.  grid = (2, 2, BB*KSPLIT), block = 256 (8 waves).
// Each WG: 128x128 f32 tile of energy[b] over one K segment of 2304.
// part[s][b][c][d] = sum_{k in seg s} Z1[b,c,k] * Zr[b,d,k]
// ---------------------------------------------------------------------------
#define S1_STRIDE 72   // 64 k + 8 pad halves, 144B rows (16B aligned)

__global__ __launch_bounds__(256)
void energy_gemm_kernel(const float* __restrict__ Z1, const float* __restrict__ Zr,
                        float* __restrict__ part) {
  __shared__ _Float16 lA[128 * S1_STRIDE];
  __shared__ _Float16 lB[128 * S1_STRIDE];

  const int mblk = blockIdx.x, nblk = blockIdx.y;
  const int b = blockIdx.z >> 2, s = blockIdx.z & 3;
  const int m0 = mblk * 128, n0 = nblk * 128;
  const int t = threadIdx.x;
  const int wave = t >> 5, lane = t & 31;

  const float* __restrict__ qb = Z1 + (size_t)b * CC * NN;
  const float* __restrict__ kb = Zr + (size_t)b * CC * NN;

  v8f acc[8] = {};

  const int k_beg = s * KSEG, k_end = k_beg + KSEG;
  for (int k0 = k_beg; k0 < k_end; k0 += 64) {
    // ---- prefetch next chunk one iteration ahead (global_prefetch_b8) ----
    if (k0 + 64 < k_end) {
      const int prow = t >> 1, pcol = (t & 1) * 32;
      __builtin_prefetch(qb + (size_t)(m0 + prow) * NN + k0 + 64 + pcol, 0, 0);
      __builtin_prefetch(kb + (size_t)(n0 + prow) * NN + k0 + 64 + pcol, 0, 0);
    }
    // ---- stage 128x64 f32 of each operand into LDS as f16 ----
#pragma unroll
    for (int i = 0; i < 8; ++i) {
      const int idx = i * 256 + t;          // 2048 float4 per buffer
      const int row = idx >> 4;             // 16 float4 per 64-wide row
      const int c4  = idx & 15;
      v4f qa = *(const v4f*)(qb + (size_t)(m0 + row) * NN + k0 + c4 * 4);
      v4f ka = *(const v4f*)(kb + (size_t)(n0 + row) * NN + k0 + c4 * 4);
      v4h qh, kh;
#pragma unroll
      for (int j = 0; j < 4; ++j) { qh[j] = (_Float16)qa[j]; kh[j] = (_Float16)ka[j]; }
      *(v4h*)(lA + row * S1_STRIDE + c4 * 4) = qh;
      *(v4h*)(lB + row * S1_STRIDE + c4 * 4) = kh;
    }
    __syncthreads();

    // ---- 2 x K=32 WMMA steps; wave owns m-tile = wave, 8 n-tiles ----
#pragma unroll
    for (int kk = 0; kk < 64; kk += 32) {
      v16h af = load_a_frag(lA + (wave * 16 + (lane & 15)) * S1_STRIDE + kk, lane);
#pragma unroll
      for (int nt = 0; nt < 8; ++nt) {
        v16h bf = load_b_frag(lB + (nt * 16 + (lane & 15)) * S1_STRIDE + kk, lane);
        acc[nt] = __builtin_amdgcn_wmma_f32_16x16x32_f16(
            false, af, false, bf, (short)0, acc[nt], false, false);
      }
    }
    __syncthreads();
  }

  // ---- write 16x16 f32 tiles: VGPR v -> M = v + 8*(lane>=16), N = lane&15 ----
  float* __restrict__ pout = part + ((size_t)(s * BB + b)) * CC * CC;
  const int mbase = m0 + wave * 16 + ((lane >> 4) << 3);
  const int col = lane & 15;
#pragma unroll
  for (int nt = 0; nt < 8; ++nt)
#pragma unroll
    for (int v = 0; v < 8; ++v)
      pout[(size_t)(mbase + v) * CC + n0 + nt * 16 + col] = acc[nt][v];
}

// ---------------------------------------------------------------------------
// Kernel 2: reduce K-split partials + softmax(min-trick) -> f16 attention.
// softmax(max_d(e) - e) over d  ==  exp(min_d(e) - e) / sum.
// One wave32 per 256-length row; 8 rows per 256-thread block.
// ---------------------------------------------------------------------------
__global__ __launch_bounds__(256)
void softmax_kernel(const float* __restrict__ part, _Float16* __restrict__ att) {
  const int wave = threadIdx.x >> 5, lane = threadIdx.x & 31;
  const int row = blockIdx.x * 8 + wave;                 // 0 .. BB*CC-1
  const size_t roff = (size_t)row * CC;
  const size_t seg = (size_t)BB * CC * CC;

  float v[8];
  float mn = 3.402823466e+38f;
#pragma unroll
  for (int i = 0; i < 8; ++i) {
    const size_t j = roff + lane + i * 32;
    v[i] = part[j] + part[seg + j] + part[2 * seg + j] + part[3 * seg + j];
    mn = fminf(mn, v[i]);
  }
#pragma unroll
  for (int off = 16; off > 0; off >>= 1) mn = fminf(mn, __shfl_xor(mn, off));

  float sum = 0.f;
#pragma unroll
  for (int i = 0; i < 8; ++i) { v[i] = __expf(mn - v[i]); sum += v[i]; }
#pragma unroll
  for (int off = 16; off > 0; off >>= 1) sum += __shfl_xor(sum, off);

  const float inv = 1.0f / sum;
#pragma unroll
  for (int i = 0; i < 8; ++i)
    att[roff + lane + i * 32] = (_Float16)(v[i] * inv);
}

// ---------------------------------------------------------------------------
// Kernel 3: out[b,c,n] = beta * sum_d att[b,c,d] * Zr[b,d,n] + Zr[b,c,n]
// grid = (NN/64, BB), block = 512 (16 waves). Wave w owns m-tile w (16 c-rows)
// x 4 n-tiles. K (= d, 256) staged in 32-wide LDS chunks; Zr slab transposed.
// Attention (128 KB/batch) rides in L2 across the 144 N-slabs; its staging is
// a pure f16 byte copy -> use async global->LDS DMA when available.
// ---------------------------------------------------------------------------
#define S3_STRIDE 40   // 32 k + 8 pad halves, 80B rows (16B aligned)

__global__ __launch_bounds__(512)
void out_gemm_kernel(const _Float16* __restrict__ att, const float* __restrict__ Zr,
                     const float* __restrict__ beta_p, float* __restrict__ out) {
  __shared__ _Float16 lA[CC * S3_STRIDE];   // att[c][d-chunk]
  __shared__ _Float16 lK[64 * S3_STRIDE];   // Zr transposed: [n_local][d-chunk]

  const int n0 = blockIdx.x * 64;
  const int b  = blockIdx.y;
  const int t = threadIdx.x;
  const int wave = t >> 5, lane = t & 31;

  const _Float16* __restrict__ ab  = att + (size_t)b * CC * CC;
  const float*    __restrict__ zrb = Zr  + (size_t)b * CC * NN;

  v8f acc[4] = {};

  for (int d0 = 0; d0 < CC; d0 += 32) {
    // ---- stage attention chunk: 256 rows x 32 halves (16B per lane) ----
    {
      const int a4 = t & 3, row = t >> 2;   // 0..127
#pragma unroll
      for (int r = 0; r < 2; ++r) {
        const int rr = row + r * 128;
        const _Float16* gsrc = ab + (size_t)rr * CC + d0 + a4 * 8;
        _Float16* ldst = lA + rr * S3_STRIDE + a4 * 8;
#if HAVE_ASYNC_LDS
        __builtin_amdgcn_global_load_async_to_lds_b128(
            (gv4i*)gsrc, (lv4i*)ldst, 0, 0);
#else
        *(v8h*)ldst = *(const v8h*)gsrc;
#endif
      }
    }
    // ---- stage Zr chunk transposed: 32 d-rows x 64 n, f32 -> f16 ----
    {
      const int n = t & 63, dl0 = t >> 6;   // 0..7
#pragma unroll
      for (int r = 0; r < 4; ++r) {
        const int dl = dl0 + r * 8;
        const float x = zrb[(size_t)(d0 + dl) * NN + n0 + n];
        lK[n * S3_STRIDE + dl] = (_Float16)x;
      }
    }
    wait_async_zero();
    __syncthreads();

    v16h af = load_a_frag(lA + (wave * 16 + (lane & 15)) * S3_STRIDE, lane);
#pragma unroll
    for (int nt = 0; nt < 4; ++nt) {
      v16h bf = load_b_frag(lK + (nt * 16 + (lane & 15)) * S3_STRIDE, lane);
      acc[nt] = __builtin_amdgcn_wmma_f32_16x16x32_f16(
          false, af, false, bf, (short)0, acc[nt], false, false);
    }
    __syncthreads();
  }

  // ---- fused epilogue: beta * acc + Zr[b,c,n] ----
  const float beta = beta_p[0];
  const int cbase = wave * 16 + ((lane >> 4) << 3);
  const int col = lane & 15;
  float* __restrict__ ob = out + (size_t)b * CC * NN;
#pragma unroll
  for (int nt = 0; nt < 4; ++nt)
#pragma unroll
    for (int v = 0; v < 8; ++v) {
      const int c = cbase + v;
      const int n = n0 + nt * 16 + col;
      ob[(size_t)c * NN + n] = beta * acc[nt][v] + zrb[(size_t)c * NN + n];
    }
}

// ---------------------------------------------------------------------------
extern "C" void kernel_launch(void* const* d_in, const int* in_sizes, int n_in,
                              void* d_out, int out_size, void* d_ws, size_t ws_size,
                              hipStream_t stream) {
  const float* Z1   = (const float*)d_in[0];
  const float* Zr   = (const float*)d_in[1];
  const float* beta = (const float*)d_in[2];
  float* out = (float*)d_out;

  // d_ws layout: [KSPLIT][B][C][C] f32 energy partials (16 MB) + f16 attention (2 MB)
  float* part = (float*)d_ws;
  _Float16* att = (_Float16*)((char*)d_ws +
                              (size_t)KSPLIT * BB * CC * CC * sizeof(float));

  energy_gemm_kernel<<<dim3(2, 2, BB * KSPLIT), 256, 0, stream>>>(Z1, Zr, part);
  softmax_kernel<<<(BB * CC) / 8, 256, 0, stream>>>(part, att);
  out_gemm_kernel<<<dim3(NN / 64, BB), 512, 0, stream>>>(att, Zr, beta, out);
}
